// NeuromorphicSRWKVTpu_28552942584257
// MI455X (gfx1250) — matrix operand order repaired and checked by reading, not compiled, not run on hardware
//
#include <hip/hip_runtime.h>
#include <cstdint>
#include <cstddef>

// ---------------- problem constants (from the reference) ----------------
#define BB   2
#define TT   2048
#define DDIM 1024
#define HHN  16
#define DH   64
#define VV   32000
#define MR   (BB * TT)          // 4096 flattened rows

// ---------------- types ----------------
typedef unsigned int u32;
typedef unsigned long long u64;
typedef __attribute__((ext_vector_type(4)))  u32     u32x4;
typedef __attribute__((ext_vector_type(16))) __bf16  v16bf;
typedef __attribute__((ext_vector_type(8)))  float   v8f;

union BFPack { u32x4 q[2]; v16bf v; };
union QPack  { u32x4 q; __bf16 e[8]; };

static __device__ inline v16bf ld_pair(const __bf16* p0, const __bf16* p1) {
  BFPack t;
  t.q[0] = *(const u32x4*)p0;
  t.q[1] = *(const u32x4*)p1;
  return t.v;
}

#define WMMA_BF16(a, b, c) \
  __builtin_amdgcn_wmma_f32_16x16x32_bf16(false, (a), false, (b), (short)0, (c), false, false)

// Async 16B copy global -> LDS (GV mode, per-lane addresses). Tracked by
// ASYNCcnt; the "memory" clobber orders it against later LDS fragment loads.
static __device__ inline void async_copy_b128(unsigned lds_off, const void* gptr) {
  asm volatile("global_load_async_to_lds_b128 %0, %1, off"
               :: "v"(lds_off), "v"((u64)(uintptr_t)gptr) : "memory");
}

// =====================================================================
// Kernel 1: spiking k-WTA over token stream -> gains[DDIM]
// Every step the touched slot fires (pot>=0, then +1.0 >= thr), so
// spk == bincount and pot is decayed roundoff residue; reproduce with
// lazy per-slot decay. 1024 threads x 32 vocab slots each.
// =====================================================================
__global__ __launch_bounds__(1024) void kwta_kernel(const int* __restrict__ tok,
                                                    float* __restrict__ gains) {
  __shared__ int   sIds[MR];
  __shared__ float sV[1024];
  __shared__ int   sI[1024];
  __shared__ int   chosen[5];
  const int tid = threadIdx.x;

  for (int i = tid; i < MR; i += 1024) sIds[i] = tok[i];
  __syncthreads();

  float pot[32]; int spk[32]; int last[32];
#pragma unroll
  for (int j = 0; j < 32; ++j) { pot[j] = 0.f; spk[j] = 0; last[j] = -1; }

  const int base = tid << 5;
  for (int s = 0; s < MR; ++s) {
    int id = sIds[s];                 // LDS broadcast read
    if ((id >> 5) == tid) {
      int j = id & 31;
      float p = pot[j] * __powf(0.7f, (float)(s - last[j]));
      p = (p + 1.0f) - 1.0f;          // fire + soft reset (keeps fp residue)
      pot[j] = p; spk[j]++; last[j] = s;
    }
  }

  float vals[32];
#pragma unroll
  for (int j = 0; j < 32; ++j) {
    int idx = base + j;
    float pf = pot[j] * __powf(0.7f, (float)(MR - 1 - last[j]));
    vals[j] = (idx < VV) ? ((float)spk[j] * 1e6f + pf) : -1.0f;
  }

  // top-5 (max value, ties -> lowest index) via 5 argmax reductions
  for (int r = 0; r < 5; ++r) {
    float bv = -2.0f; int bi = 0x7fffffff;
    for (int j = 0; j < 32; ++j) {
      int idx = base + j;
      bool taken = false;
      for (int c = 0; c < r; ++c) taken = taken || (chosen[c] == idx);
      if (taken) continue;
      float v = vals[j];
      if (v > bv || (v == bv && idx < bi)) { bv = v; bi = idx; }
    }
    sV[tid] = bv; sI[tid] = bi;
    __syncthreads();
    for (int off = 512; off > 0; off >>= 1) {
      if (tid < off) {
        float ov = sV[tid + off]; int oi = sI[tid + off];
        if (ov > sV[tid] || (ov == sV[tid] && oi < sI[tid])) { sV[tid] = ov; sI[tid] = oi; }
      }
      __syncthreads();
    }
    if (tid == 0) chosen[r] = sI[0];
    __syncthreads();
  }

  if (base < DDIM) {
    for (int j = 0; j < 32; ++j) {
      int idx = base + j;
      float g = (spk[j] > 0) ? 0.6f : 1.0f;
      for (int c = 0; c < 5; ++c) if (chosen[c] == idx) g = 1.5f;
      gains[idx] = g;
    }
  }
}

// =====================================================================
// Kernel 2: fp32 -> bf16 weight conversion
// =====================================================================
__global__ void cvt_bf16_kernel(const float* __restrict__ src, __bf16* __restrict__ dst, int n) {
  int i = blockIdx.x * 256 + threadIdx.x;
  if (i < n) dst[i] = (__bf16)src[i];
}

// =====================================================================
// Kernel 3: token shift + adaptive time-mix -> xr/xk/xv (bf16)
// =====================================================================
__global__ void mix_kernel(const float* __restrict__ x, const float* __restrict__ gains,
                           const float* __restrict__ tmr, const float* __restrict__ tmk,
                           const float* __restrict__ tmv,
                           __bf16* __restrict__ xr, __bf16* __restrict__ xk,
                           __bf16* __restrict__ xv) {
  int i = blockIdx.x * 256 + threadIdx.x;
  if (i >= BB * TT * DDIM) return;
  int d = i & (DDIM - 1);
  int t = (i >> 10) & (TT - 1);
  float xc = x[i];
  float px = (t == 0) ? 0.0f : x[i - DDIM];
  float g  = gains[d];
  float mr = tmr[d] * g, mk = tmk[d] * g, mv = tmv[d] * g;
  xr[i] = (__bf16)(mr * xc + (1.0f - mr) * px);
  xk[i] = (__bf16)(mk * xc + (1.0f - mk) * px);
  xv[i] = (__bf16)(mv * xc + (1.0f - mv) * px);
}

// =====================================================================
// Kernel 4: C[M,N] = A[M,K] @ W[N,K]^T via v_wmma_f32_16x16x32_bf16.
// Both operands stream from global in native WMMA fragment layouts:
//  A 16x32:  lane l -> row l%16; VGPR0-3 = K (8*half .. +8), VGPR4-7 = +16
//  B 32x16:  lane l -> col l%16; VGPR0-3 = K (8*half .. +8), VGPR4-7 = +16
// Plain SSA loop body + unroll 2: the scheduler overlaps next-step
// loads with this step's WMMAs without register-copy chains.
// act: 0 = none, 1 = spike ((v>0.5)?1:0).
// outBT: optional per-head transposed bf16 store  vT[b][h][dh][t]
//        (C-layout holds 8 consecutive token rows per lane -> one 16B store)
// =====================================================================
__global__ __launch_bounds__(256) void gemm_bf16_kernel(
    const __bf16* __restrict__ A, const __bf16* __restrict__ W,
    float* __restrict__ outF, __bf16* __restrict__ outB,
    __bf16* __restrict__ outBT,
    const float* __restrict__ bias, int act) {
  const int K = DDIM, N = DDIM;
  const int wave = threadIdx.x >> 5, lane = threadIdx.x & 31;
  const int half = lane >> 4, lm = lane & 15;
  const int m0 = blockIdx.x * 128 + wave * 16;
  const int n0 = blockIdx.y * 64;

  v8f acc[4] = {};
  const __bf16* aRow = A + (size_t)(m0 + lm) * K + 8 * half;
  const __bf16* wp[4];
#pragma unroll
  for (int nt = 0; nt < 4; ++nt)
    wp[nt] = W + (size_t)(n0 + 16 * nt + lm) * K + 8 * half;

#pragma unroll 2
  for (int k0 = 0; k0 < K; k0 += 32) {
    v16bf a = ld_pair(aRow + k0, aRow + k0 + 16);
#pragma unroll
    for (int nt = 0; nt < 4; ++nt) {
      v16bf b = ld_pair(wp[nt] + k0, wp[nt] + k0 + 16);
      acc[nt] = WMMA_BF16(a, b, acc[nt]);
    }
  }

#pragma unroll
  for (int nt = 0; nt < 4; ++nt) {
    int col = n0 + 16 * nt + lm;
    float bv = bias ? bias[col] : 0.0f;
#pragma unroll
    for (int j = 0; j < 8; ++j) {
      float v = acc[nt][j] + bv;
      if (act == 1) v = (v > 0.5f) ? 1.0f : 0.0f;
      acc[nt][j] = v;
    }
    if (outF || outB) {
#pragma unroll
      for (int j = 0; j < 8; ++j) {
        int row = m0 + j + 8 * half;
        size_t o = (size_t)row * N + col;
        if (outF) outF[o] = acc[nt][j];
        if (outB) outB[o] = (__bf16)acc[nt][j];
      }
    }
    if (outBT) {
      QPack pk;
#pragma unroll
      for (int j = 0; j < 8; ++j) pk.e[j] = (__bf16)acc[nt][j];
      int row0 = m0 + 8 * half;             // 8 consecutive token rows
      int bidx = row0 >> 11;                // row0 / TT
      int t0   = row0 & (TT - 1);
      int hh = col >> 6, dh = col & (DH - 1);
      size_t o = ((size_t)(bidx * HHN + hh) * DH + dh) * TT + t0;
      *(u32x4*)(outBT + o) = pk.q;          // t0 % 8 == 0 -> 16B aligned
    }
  }
}

// =====================================================================
// Kernel 5: flash attention (full, non-causal) + fused sigmoid(r) gating.
// Block = one (b,h) x 128 queries, 8 waves x 16 query rows.
// K (32x64) and V^T (64x32) chunks are staged into double-buffered LDS
// with global_load_async_to_lds_b128: chunk i+1's copies are issued
// before computing chunk i, then s_wait_asynccnt 0x2 retires only the
// older pair (ASYNCcnt completes in order). Per chunk: S = QK^T
// (4 wmma), online softmax via shfl_xor inside 16-lane halves, P
// bounced through per-wave LDS to A-layout, O += P*V (4 wmma).
// =====================================================================
__global__ __launch_bounds__(256) void attn_kernel(
    const __bf16* __restrict__ rb, const __bf16* __restrict__ kb,
    const __bf16* __restrict__ vT, const float* __restrict__ rf,
    __bf16* __restrict__ gout) {
  __shared__ __bf16 Kl[2][32][72];   // keys x d, rows 144B (bank stride 36)
  __shared__ __bf16 Vl[2][64][40];   // d x keys, rows  80B (bank stride 20)
  __shared__ __bf16 Ps[8][16][40];   // per-wave P bounce buffer

  const int wave = threadIdx.x >> 5, lane = threadIdx.x & 31;
  const int half = lane >> 4, lm = lane & 15;
  const int h = blockIdx.y, b = blockIdx.z;
  const size_t rowbase = (size_t)b * TT;
  const int hc = h * DH;
  const int q0 = blockIdx.x * 128 + wave * 16;
  const __bf16* vhead = vT + (size_t)(b * HHN + h) * DH * TT;

  // Q rows as A-operands (Dh=64 -> two K=32 operands), kept in registers.
  const __bf16* qp = rb + (rowbase + q0 + lm) * DDIM + hc + 8 * half;
  v16bf Aq0 = ld_pair(qp, qp + 16);
  v16bf Aq1 = ld_pair(qp + 32, qp + 48);

  // per-thread staging assignments (256 threads, one 16B async copy each)
  const int tk_key = threadIdx.x >> 3, tk_seg = threadIdx.x & 7;  // K: 32x8
  const int tv_row = threadIdx.x >> 2, tv_seg = threadIdx.x & 3;  // V: 64x4
  unsigned kl_dst[2], vl_dst[2];
#pragma unroll
  for (int s = 0; s < 2; ++s) {
    kl_dst[s] = (unsigned)(uintptr_t)&Kl[s][tk_key][tk_seg * 8];
    vl_dst[s] = (unsigned)(uintptr_t)&Vl[s][tv_row][tv_seg * 8];
  }
  const __bf16* ksrc = kb + (rowbase + tk_key) * DDIM + hc + tk_seg * 8;
  const __bf16* vsrc = vhead + (size_t)tv_row * TT + tv_seg * 8;

  float mrow[8], lrow[8];
  v8f Oa[4] = {};
#pragma unroll
  for (int j = 0; j < 8; ++j) { mrow[j] = -1e30f; lrow[j] = 0.f; }

  const int NCH = TT / 32;             // 64 key chunks
  // prologue: stage chunk 0 into buffer 0
  async_copy_b128(kl_dst[0], ksrc);
  async_copy_b128(vl_dst[0], vsrc);

  for (int ic = 0; ic < NCH; ++ic) {
    const int cur = ic & 1;
    if (ic + 1 < NCH) {
      // issue next chunk into the other buffer, then retire only the
      // older pair of copies (the 2 newest may remain outstanding)
      int kn = (ic + 1) * 32;
      async_copy_b128(kl_dst[cur ^ 1], ksrc + (size_t)kn * DDIM);
      async_copy_b128(vl_dst[cur ^ 1], vsrc + kn);
      asm volatile("s_wait_asynccnt 0x2" ::: "memory");
    } else {
      asm volatile("s_wait_asynccnt 0x0" ::: "memory");
    }
    __syncthreads();

    // ---- S = Q K^T : B-operand = K rows from LDS (contiguous K-dim) ----
    v8f C1 = {}, C2 = {};
    {
      const __bf16* p1 = &Kl[cur][lm][8 * half];
      v16bf b00 = ld_pair(p1, p1 + 16);
      v16bf b01 = ld_pair(p1 + 32, p1 + 48);
      C1 = WMMA_BF16(Aq0, b00, C1);
      C1 = WMMA_BF16(Aq1, b01, C1);
      const __bf16* p2 = &Kl[cur][16 + lm][8 * half];
      v16bf b10 = ld_pair(p2, p2 + 16);
      v16bf b11 = ld_pair(p2 + 32, p2 + 48);
      C2 = WMMA_BF16(Aq0, b10, C2);
      C2 = WMMA_BF16(Aq1, b11, C2);
    }

    // ---- online softmax (row j lives across the 16 lanes of this half) ----
    float p1a[8], p2a[8], alpha[8];
#pragma unroll
    for (int j = 0; j < 8; ++j) {
      float c1 = C1[j] * 0.125f, c2 = C2[j] * 0.125f;
      float cm = fmaxf(c1, c2);
#pragma unroll
      for (int mk = 1; mk < 16; mk <<= 1) cm = fmaxf(cm, __shfl_xor(cm, mk, 32));
      float nm = fmaxf(mrow[j], cm);
      float al = __expf(mrow[j] - nm);
      float e1 = __expf(c1 - nm), e2 = __expf(c2 - nm);
      float rs = e1 + e2;
#pragma unroll
      for (int mk = 1; mk < 16; mk <<= 1) rs += __shfl_xor(rs, mk, 32);
      lrow[j] = lrow[j] * al + rs;
      mrow[j] = nm;
      alpha[j] = al;
      p1a[j] = e1; p2a[j] = e2;
    }
#pragma unroll
    for (int nt = 0; nt < 4; ++nt)
#pragma unroll
      for (int j = 0; j < 8; ++j) Oa[nt][j] *= alpha[j];

    // ---- P: C-layout -> per-wave LDS -> A-layout (DS in-order per wave) ----
#pragma unroll
    for (int j = 0; j < 8; ++j) {
      int row = j + 8 * half;
      Ps[wave][row][lm]      = (__bf16)p1a[j];
      Ps[wave][row][16 + lm] = (__bf16)p2a[j];
    }
    const __bf16* pp = &Ps[wave][lm][8 * half];
    v16bf Ap = ld_pair(pp, pp + 16);

    // ---- O += P * V : B-operand = V^T rows from LDS (contiguous keys) ----
#pragma unroll
    for (int nt = 0; nt < 4; ++nt) {
      const __bf16* pv = &Vl[cur][16 * nt + lm][8 * half];
      v16bf Bv = ld_pair(pv, pv + 16);
      Oa[nt] = WMMA_BF16(Ap, Bv, Oa[nt]);
    }
    __syncthreads();   // all waves done with buf[cur] before it is re-staged
  }

  // ---- epilogue: normalize + fused receptance gate -> g (bf16) ----
#pragma unroll
  for (int nt = 0; nt < 4; ++nt)
#pragma unroll
    for (int j = 0; j < 8; ++j) {
      int q = q0 + j + 8 * half;
      int dcol = hc + 16 * nt + lm;
      size_t o = (rowbase + q) * (size_t)DDIM + dcol;
      float ov = Oa[nt][j] / lrow[j];
      float rv = rf[o];
      float g = ov / (1.0f + __expf(-rv));
      gout[o] = (__bf16)g;
    }
}

// =====================================================================
// Host-side orchestration
// =====================================================================
extern "C" void kernel_launch(void* const* d_in, const int* in_sizes, int n_in,
                              void* d_out, int out_size, void* d_ws, size_t ws_size,
                              hipStream_t stream) {
  const float* x   = (const float*)d_in[0];
  const int*   tok = (const int*)d_in[1];
  const float* Wr  = (const float*)d_in[2];
  const float* Wk  = (const float*)d_in[3];
  const float* Wv  = (const float*)d_in[4];
  const float* Wo  = (const float*)d_in[5];
  const float* bo  = (const float*)d_in[6];
  const float* tmk = (const float*)d_in[7];
  const float* tmv = (const float*)d_in[8];
  const float* tmr = (const float*)d_in[9];
  float* out = (float*)d_out;

  // workspace carve-up (~84 MB total)
  char* w = (char*)d_ws;
  auto alloc = [&](size_t bytes) -> char* {
    char* p = w;
    w += (bytes + 255) & ~(size_t)255;
    return p;
  };
  const size_t WB = (size_t)DDIM * DDIM * sizeof(__bf16);   // 2 MB
  const size_t MB = (size_t)MR * DDIM * sizeof(__bf16);     // 8 MB
  const size_t MF = (size_t)MR * DDIM * sizeof(float);      // 16 MB

  float*  gains = (float*)alloc(DDIM * sizeof(float));
  __bf16* Wrb = (__bf16*)alloc(WB);
  __bf16* Wkb = (__bf16*)alloc(WB);
  __bf16* Wvb = (__bf16*)alloc(WB);
  __bf16* Wob = (__bf16*)alloc(WB);
  __bf16* xr  = (__bf16*)alloc(MB);
  __bf16* xk  = (__bf16*)alloc(MB);
  __bf16* xv  = (__bf16*)alloc(MB);
  __bf16* rbf = (__bf16*)alloc(MB);
  __bf16* kbf = (__bf16*)alloc(MB);
  __bf16* vTb = (__bf16*)alloc(MB);   // per-head transposed V
  float*  rf  = (float*)alloc(MF);
  __bf16* gbf = (__bf16*)alloc(MB);

  // 1) kWTA gains
  kwta_kernel<<<1, 1024, 0, stream>>>(tok, gains);

  // 2) weight conversion
  const int nW = DDIM * DDIM;
  cvt_bf16_kernel<<<(nW + 255) / 256, 256, 0, stream>>>(Wr, Wrb, nW);
  cvt_bf16_kernel<<<(nW + 255) / 256, 256, 0, stream>>>(Wk, Wkb, nW);
  cvt_bf16_kernel<<<(nW + 255) / 256, 256, 0, stream>>>(Wv, Wvb, nW);
  cvt_bf16_kernel<<<(nW + 255) / 256, 256, 0, stream>>>(Wo, Wob, nW);

  // 3) token shift + time mix
  const int nE = BB * TT * DDIM;
  mix_kernel<<<(nE + 255) / 256, 256, 0, stream>>>(x, gains, tmr, tmk, tmv, xr, xk, xv);

  // 4) projections (WMMA GEMMs)
  dim3 gg(MR / 128, DDIM / 64);
  gemm_bf16_kernel<<<gg, 256, 0, stream>>>(xr, Wrb, rf, rbf, nullptr, nullptr, 0);      // r
  gemm_bf16_kernel<<<gg, 256, 0, stream>>>(xk, Wkb, nullptr, kbf, nullptr, nullptr, 1); // k = spike
  gemm_bf16_kernel<<<gg, 256, 0, stream>>>(xv, Wvb, nullptr, nullptr, vTb, nullptr, 0); // v (transposed)

  // 5) flash attention + fused sigmoid(r) gating
  dim3 ga(TT / 128, HHN, BB);
  attn_kernel<<<ga, 256, 0, stream>>>(rbf, kbf, vTb, rf, gbf);

  // 6) output projection + bias
  gemm_bf16_kernel<<<gg, 256, 0, stream>>>(gbf, Wob, out, nullptr, nullptr, bo, 0);
}
